// DeformFusion_40853728920100
// MI455X (gfx1250) — matrix operand order repaired and checked by reading, not compiled, not run on hardware
//
#include <hip/hip_runtime.h>
#include <hip/hip_bf16.h>

typedef __attribute__((ext_vector_type(16))) _Float16 v16h;
typedef __attribute__((ext_vector_type(8)))  float    v8f;

#define B_   4
#define N_   5
#define H0   128
#define W0   256
#define HW0  (H0*W0)
#define C0_  64
#define IC1  128          // 2*C0
#define OC1  32
#define K1   (IC1*9)      // 1152
#define OC2  16
#define K2   (OC1*9)      // 288
#define IMGS (B_*(N_-1))  // 16

__device__ __forceinline__ int iclamp(int v, int lo, int hi) {
    return v < lo ? lo : (v > hi ? hi : v);
}

// ---------------- weight pack: w[oc][ic][ky][kx] -> Wp[tap][oc][ic], f16 ----------------
__global__ void k_pack(const float* __restrict__ w, _Float16* __restrict__ wp, int OC, int IC) {
    int idx = blockIdx.x * blockDim.x + threadIdx.x;
    int total = OC * IC * 9;
    if (idx >= total) return;
    int ic = idx % IC;
    int t2 = idx / IC;
    int oc = t2 % OC;
    int t  = t2 / OC;
    wp[idx] = (_Float16)w[(oc * IC + ic) * 9 + t];
}

// ---------------- affine warp (grid_sample, align_corners=True) ----------------
// one thread per (bn, y, x); loops channels. Optionally emits an f16 copy (conv input).
__global__ void k_warp(const float* __restrict__ feat, const float* __restrict__ pm,
                       float* __restrict__ out, _Float16* __restrict__ outh,
                       int C, int H, int W) {
    int idx = blockIdx.x * blockDim.x + threadIdx.x;
    int total = B_ * N_ * H * W;
    if (idx >= total) return;
    int x = idx % W; int t = idx / W;
    int y = t % H;  int bn = t / H;
    int b = bn / N_; int n = bn % N_;
    const float* th = pm + ((size_t)b * N_ * N_ + n) * 6;   // pairwise[b][0][n]
    float xs = -1.f + 2.f * (float)x / (float)(W - 1);
    float ys = -1.f + 2.f * (float)y / (float)(H - 1);
    float gx = th[0] * xs + th[1] * ys + th[2];
    float gy = th[3] * xs + th[4] * ys + th[5];
    float ix = (gx + 1.f) * 0.5f * (float)(W - 1);
    float iy = (gy + 1.f) * 0.5f * (float)(H - 1);
    float x0f = floorf(ix), y0f = floorf(iy);
    int x0 = (int)x0f, y0 = (int)y0f;
    float wx1 = ix - x0f, wx0 = 1.f - wx1;
    float wy1 = iy - y0f, wy0 = 1.f - wy1;
    int xc0 = iclamp(x0, 0, W - 1), xc1 = iclamp(x0 + 1, 0, W - 1);
    int yc0 = iclamp(y0, 0, H - 1), yc1 = iclamp(y0 + 1, 0, H - 1);
    float v00 = (x0   >= 0 && x0   < W && y0   >= 0 && y0   < H) ? 1.f : 0.f;
    float v01 = (x0+1 >= 0 && x0+1 < W && y0   >= 0 && y0   < H) ? 1.f : 0.f;
    float v10 = (x0   >= 0 && x0   < W && y0+1 >= 0 && y0+1 < H) ? 1.f : 0.f;
    float v11 = (x0+1 >= 0 && x0+1 < W && y0+1 >= 0 && y0+1 < H) ? 1.f : 0.f;
    float w00 = wx0 * wy0 * v00, w01 = wx1 * wy0 * v01;
    float w10 = wx0 * wy1 * v10, w11 = wx1 * wy1 * v11;
    size_t o00 = (size_t)yc0 * W + xc0, o01 = (size_t)yc0 * W + xc1;
    size_t o10 = (size_t)yc1 * W + xc0, o11 = (size_t)yc1 * W + xc1;
    const float* ib = feat + (size_t)bn * C * H * W;
    size_t obase = (size_t)bn * C * H * W + (size_t)y * W + x;
    for (int c = 0; c < C; c++) {
        const float* p = ib + (size_t)c * H * W;
        float r = p[o00]*w00 + p[o01]*w01 + p[o10]*w10 + p[o11]*w11;
        out[obase + (size_t)c * H * W] = r;
        if (outh) outh[obase + (size_t)c * H * W] = (_Float16)r;
    }
}

// ---------------- conv1: implicit GEMM 128ch x 3x3 -> 32ch via WMMA ----------------
// K permuted as k' = tap*128 + ic. Block = 128 threads (4 waves); each wave: 32x16 tile.
__global__ __launch_bounds__(128) void k_conv1_wmma(const _Float16* __restrict__ w0h,
                                                    const _Float16* __restrict__ Wp,
                                                    float* __restrict__ y1) {
    __shared__ __align__(16) _Float16 As[OC1 * IC1];   // 32 oc x 128 ic tap slice (8KB)
    int tid  = threadIdx.x;
    int lane = tid & 31;
    int wv   = tid >> 5;
    int n0   = blockIdx.x * 64 + wv * 16;   // never straddles images (HW0 % 64 == 0)
    int img  = n0 / HW0;
    int pix0 = n0 % HW0;
    int b = img >> 2;
    int a = img & 3;
    const _Float16* egoBase = w0h + (size_t)(b * N_)         * C0_ * HW0;
    const _Float16* neiBase = w0h + (size_t)(b * N_ + a + 1) * C0_ * HW0;
    int nl   = lane & 15;
    int half = lane >> 4;
    int pix  = pix0 + nl;
    int py = pix >> 8;        // W0 == 256
    int px = pix & 255;

    v8f acc0 = {};
    v8f acc1 = {};
    for (int t = 0; t < 9; t++) {
        // stage tap slice: 4096 halfs, 128 threads x 4 uint4 (b128 loads/stores)
        {
            const uint4* s4 = (const uint4*)(Wp + (size_t)t * OC1 * IC1);
            uint4* d4 = (uint4*)As;
#pragma unroll
            for (int i = 0; i < 4; i++) d4[tid * 4 + i] = s4[tid * 4 + i];
        }
        __syncthreads();
        int ky = t / 3, kx = t - 3 * (t / 3);
        int yy = py + ky - 1, xx = px + kx - 1;
        bool inb = (yy >= 0 && yy < H0 && xx >= 0 && xx < W0);
        // clamp so the address is always valid; zero out-of-bounds via select (no branches)
        size_t soff = (size_t)iclamp(yy, 0, H0 - 1) * W0 + iclamp(xx, 0, W0 - 1);
        const _Float16 hz = (_Float16)0.f;
#pragma unroll
        for (int cb = 0; cb < IC1; cb += 32) {
            const _Float16* srcB = (cb < C0_) ? (egoBase + (size_t)cb * HW0 + soff)
                                              : (neiBase + (size_t)(cb - C0_) * HW0 + soff);
            v16h bf;
#pragma unroll
            for (int v = 0; v < 8; v++) {
                int ic = half * 16 + 2 * v;
                _Float16 f0 = srcB[(size_t)ic * HW0];
                _Float16 f1 = srcB[(size_t)(ic + 1) * HW0];
                bf[2 * v]     = inb ? f0 : hz;
                bf[2 * v + 1] = inb ? f1 : hz;
            }
            v16h a0, a1;
#pragma unroll
            for (int v = 0; v < 8; v++) {
                int kk = cb + 16 * half + 2 * v;
                a0[2*v]   = As[nl * IC1 + kk];
                a0[2*v+1] = As[nl * IC1 + kk + 1];
                a1[2*v]   = As[(16 + nl) * IC1 + kk];
                a1[2*v+1] = As[(16 + nl) * IC1 + kk + 1];
            }
            acc0 = __builtin_amdgcn_wmma_f32_16x16x32_f16(false, a0, false, bf, (short)0, acc0, false, false);
            acc1 = __builtin_amdgcn_wmma_f32_16x16x32_f16(false, a1, false, bf, (short)0, acc1, false, false);
        }
        __syncthreads();
    }
    // D layout: VGPR r -> M = r + 8*half, N = nl  (bias b1 cancels in instance-norm)
#pragma unroll
    for (int r = 0; r < 8; r++) {
        int m0 = r + 8 * half;
        y1[((size_t)img * OC1 + m0)      * HW0 + pix] = acc0[r];
        y1[((size_t)img * OC1 + 16 + m0) * HW0 + pix] = acc1[r];
    }
}

// ---------------- conv2: implicit GEMM 32ch x 3x3 -> 16ch via WMMA ----------------
__global__ __launch_bounds__(128) void k_conv2_wmma(const _Float16* __restrict__ y1h,
                                                    const _Float16* __restrict__ Wp,
                                                    float* __restrict__ y2) {
    __shared__ __align__(16) _Float16 As[OC2 * 32];    // 16 oc x 32 ic tap slice (1KB)
    int tid  = threadIdx.x;
    int lane = tid & 31;
    int wv   = tid >> 5;
    int n0   = blockIdx.x * 64 + wv * 16;
    int img  = n0 / HW0;
    int pix0 = n0 % HW0;
    const _Float16* inBase = y1h + (size_t)img * OC1 * HW0;
    int nl   = lane & 15;
    int half = lane >> 4;
    int pix  = pix0 + nl;
    int py = pix >> 8;
    int px = pix & 255;

    v8f acc = {};
    for (int t = 0; t < 9; t++) {
        // stage 512 halfs: 128 threads x 1 uint2
        {
            const uint2* s2 = (const uint2*)(Wp + (size_t)t * OC2 * 32);
            uint2* d2 = (uint2*)As;
            d2[tid] = s2[tid];
        }
        __syncthreads();
        int ky = t / 3, kx = t - 3 * (t / 3);
        int yy = py + ky - 1, xx = px + kx - 1;
        bool inb = (yy >= 0 && yy < H0 && xx >= 0 && xx < W0);
        size_t soff = (size_t)iclamp(yy, 0, H0 - 1) * W0 + iclamp(xx, 0, W0 - 1);
        const _Float16* srcB = inBase + soff;
        const _Float16 hz = (_Float16)0.f;
        v16h bf;
#pragma unroll
        for (int v = 0; v < 8; v++) {
            int ic = half * 16 + 2 * v;
            _Float16 f0 = srcB[(size_t)ic * HW0];
            _Float16 f1 = srcB[(size_t)(ic + 1) * HW0];
            bf[2 * v]     = inb ? f0 : hz;
            bf[2 * v + 1] = inb ? f1 : hz;
        }
        v16h a0;
#pragma unroll
        for (int v = 0; v < 8; v++) {
            int kk = 16 * half + 2 * v;
            a0[2*v]   = As[nl * 32 + kk];
            a0[2*v+1] = As[nl * 32 + kk + 1];
        }
        acc = __builtin_amdgcn_wmma_f32_16x16x32_f16(false, a0, false, bf, (short)0, acc, false, false);
        __syncthreads();
    }
#pragma unroll
    for (int r = 0; r < 8; r++) {
        int m0 = r + 8 * half;
        y2[((size_t)img * OC2 + m0) * HW0 + pix] = acc[r];
    }
}

// ---------------- instance norm (+ leaky) in place; optional f16 copy ----------------
__global__ __launch_bounds__(256) void k_instnorm(float* __restrict__ x, _Float16* __restrict__ xh,
                                                  int HW, int do_leaky) {
    float* p = x + (size_t)blockIdx.x * HW;
    __shared__ float ssum[256], ssq[256];
    int tid = threadIdx.x;
    float s = 0.f, q = 0.f;
    for (int i = tid; i < HW; i += 256) { float v = p[i]; s += v; q += v * v; }
    ssum[tid] = s; ssq[tid] = q;
    __syncthreads();
    for (int off = 128; off > 0; off >>= 1) {
        if (tid < off) { ssum[tid] += ssum[tid + off]; ssq[tid] += ssq[tid + off]; }
        __syncthreads();
    }
    float mean = ssum[0] / (float)HW;
    float var  = ssq[0] / (float)HW - mean * mean;
    float inv  = rsqrtf(var + 1e-5f);
    _Float16* ph = xh ? (xh + (size_t)blockIdx.x * HW) : (_Float16*)0;
    for (int i = tid; i < HW; i += 256) {
        float v = (p[i] - mean) * inv;
        if (do_leaky) v = v >= 0.f ? v : 0.01f * v;
        p[i] = v;
        if (ph) ph[i] = (_Float16)v;
    }
}

// ---------------- conv3 1x1 16->8 (bias cancels in IN) ----------------
__global__ void k_conv3(const float* __restrict__ y2, const float* __restrict__ w3,
                        float* __restrict__ y3) {
    int idx = blockIdx.x * blockDim.x + threadIdx.x;
    if (idx >= IMGS * HW0) return;
    int img = idx / HW0, pix = idx % HW0;
    const float* in = y2 + (size_t)img * OC2 * HW0 + pix;
    float vin[16];
#pragma unroll
    for (int i = 0; i < 16; i++) vin[i] = in[(size_t)i * HW0];
#pragma unroll
    for (int o = 0; o < 8; o++) {
        float s = 0.f;
#pragma unroll
        for (int i = 0; i < 16; i++) s += w3[o * 16 + i] * vin[i];
        y3[((size_t)img * 8 + o) * HW0 + pix] = s;
    }
}

// ---------------- conv4 1x1 8->2 + b4, natural NCHW layout ----------------
__global__ void k_conv4(const float* __restrict__ y3, const float* __restrict__ w4,
                        const float* __restrict__ b4, float* __restrict__ g4) {
    int idx = blockIdx.x * blockDim.x + threadIdx.x;
    if (idx >= IMGS * HW0) return;
    int img = idx / HW0, pix = idx % HW0;
    const float* in = y3 + (size_t)img * 8 * HW0 + pix;
    float vin[8];
#pragma unroll
    for (int i = 0; i < 8; i++) vin[i] = in[(size_t)i * HW0];
#pragma unroll
    for (int o = 0; o < 2; o++) {
        float s = b4[o];
#pragma unroll
        for (int i = 0; i < 8; i++) s += w4[o * 8 + i] * vin[i];
        g4[((size_t)img * 2 + o) * HW0 + pix] = s;
    }
}

// ---------------- deform-sample neighbors + max-fuse with ego ----------------
// grid read replicates the reference's flat (16,2,H,W)->(16,H,W,2) reshape.
__global__ void k_fuse(const float* __restrict__ warp, const float* __restrict__ g4,
                       float* __restrict__ out, int C, int Hs, int Ws, int stride) {
    int idx = blockIdx.x * blockDim.x + threadIdx.x;
    int total = B_ * Hs * Ws;
    if (idx >= total) return;
    int xs = idx % Ws; int t = idx / Ws;
    int ys = t % Hs;  int b = t / Hs;
    int HWs = Hs * Ws;
    int h = ys * stride, w = xs * stride;
    float basex = ((float)w + 0.5f) * (2.0f / (float)W0) - 1.0f;
    float basey = ((float)h + 0.5f) * (2.0f / (float)H0) - 1.0f;
    size_t c00[4], c01[4], c10[4], c11[4];
    float f00[4], f01[4], f10[4], f11[4];
    for (int a = 0; a < 4; a++) {
        int img = b * 4 + a;
        size_t go = (size_t)img * 2 * HW0 + ((size_t)h * W0 + w) * 2;   // flat reshape view
        float gx = g4[go]     + basex;
        float gy = g4[go + 1] + basey;
        float ix = ((gx + 1.f) * (float)Ws - 1.f) * 0.5f;   // align_corners=False
        float iy = ((gy + 1.f) * (float)Hs - 1.f) * 0.5f;
        float x0f = floorf(ix), y0f = floorf(iy);
        int x0 = (int)x0f, y0 = (int)y0f;
        float wx1 = ix - x0f, wx0 = 1.f - wx1;
        float wy1 = iy - y0f, wy0 = 1.f - wy1;
        int xc0 = iclamp(x0, 0, Ws - 1), xc1 = iclamp(x0 + 1, 0, Ws - 1);
        int yc0 = iclamp(y0, 0, Hs - 1), yc1 = iclamp(y0 + 1, 0, Hs - 1);
        float v00 = (x0   >= 0 && x0   < Ws && y0   >= 0 && y0   < Hs) ? 1.f : 0.f;
        float v01 = (x0+1 >= 0 && x0+1 < Ws && y0   >= 0 && y0   < Hs) ? 1.f : 0.f;
        float v10 = (x0   >= 0 && x0   < Ws && y0+1 >= 0 && y0+1 < Hs) ? 1.f : 0.f;
        float v11 = (x0+1 >= 0 && x0+1 < Ws && y0+1 >= 0 && y0+1 < Hs) ? 1.f : 0.f;
        f00[a] = wx0 * wy0 * v00; f01[a] = wx1 * wy0 * v01;
        f10[a] = wx0 * wy1 * v10; f11[a] = wx1 * wy1 * v11;
        c00[a] = (size_t)yc0 * Ws + xc0; c01[a] = (size_t)yc0 * Ws + xc1;
        c10[a] = (size_t)yc1 * Ws + xc0; c11[a] = (size_t)yc1 * Ws + xc1;
    }
    const float* egoB = warp + (size_t)(b * N_) * C * HWs;
    size_t sp = (size_t)ys * Ws + xs;
    for (int c = 0; c < C; c++) {
        float m = egoB[(size_t)c * HWs + sp];
        for (int a = 0; a < 4; a++) {
            const float* p = warp + ((size_t)(b * N_ + a + 1) * C + c) * HWs;
            float sv = p[c00[a]] * f00[a] + p[c01[a]] * f01[a]
                     + p[c10[a]] * f10[a] + p[c11[a]] * f11[a];
            m = fmaxf(m, sv);
        }
        out[((size_t)b * C + c) * HWs + sp] = m;
    }
}

extern "C" void kernel_launch(void* const* d_in, const int* in_sizes, int n_in,
                              void* d_out, int out_size, void* d_ws, size_t ws_size,
                              hipStream_t stream) {
    const float* feat0 = (const float*)d_in[0];
    const float* feat1 = (const float*)d_in[1];
    const float* feat2 = (const float*)d_in[2];
    const float* pm    = (const float*)d_in[3];
    // d_in[4] record_len unused (always N)
    const float* w1 = (const float*)d_in[5];
    const float* w2 = (const float*)d_in[7];
    const float* w3 = (const float*)d_in[9];
    const float* w4 = (const float*)d_in[11];
    const float* b4 = (const float*)d_in[12];
    float* ws = (float*)d_ws;
    float* out = (float*)d_out;

    // workspace layout (float units; every offset is a multiple of 4 -> 16B aligned)
    size_t oW0  = 0;                                   // warped0 f32: 20*64*HW0
    size_t oW1  = oW0  + (size_t)20 * 64 * HW0;        // warped1 f32
    size_t oW2  = oW1  + (size_t)20 * 128 * 64 * 128;  // warped2 f32
    size_t oY1  = oW2  + (size_t)20 * 256 * 32 * 64;   // y1 f32
    size_t oY2  = oY1  + (size_t)IMGS * OC1 * HW0;     // y2 f32
    size_t oY3  = oY2  + (size_t)IMGS * OC2 * HW0;     // y3 f32
    size_t oG4  = oY3  + (size_t)IMGS * 8 * HW0;       // conv4 out f32
    size_t oW0H = oG4  + (size_t)IMGS * 2 * HW0;       // warped0 f16 (half-sized in floats)
    size_t oY1H = oW0H + (size_t)20 * 64 * HW0 / 2;    // y1 f16
    size_t oWP1 = oY1H + (size_t)IMGS * OC1 * HW0 / 2; // packed w1 f16
    size_t oWP2 = oWP1 + (size_t)(OC1 * K1) / 2;       // packed w2 f16
    float* warp0 = ws + oW0;
    float* warp1 = ws + oW1;
    float* warp2 = ws + oW2;
    float* y1 = ws + oY1;
    float* y2 = ws + oY2;
    float* y3 = ws + oY3;
    float* g4 = ws + oG4;
    _Float16* w0h = (_Float16*)(ws + oW0H);
    _Float16* y1h = (_Float16*)(ws + oY1H);
    _Float16* wp1 = (_Float16*)(ws + oWP1);
    _Float16* wp2 = (_Float16*)(ws + oWP2);

    k_pack<<<(OC1 * K1 + 255) / 256, 256, 0, stream>>>(w1, wp1, OC1, IC1);
    k_pack<<<(OC2 * K2 + 255) / 256, 256, 0, stream>>>(w2, wp2, OC2, OC1);

    k_warp<<<(20 * 128 * 256 + 255) / 256, 256, 0, stream>>>(feat0, pm, warp0, w0h, 64, 128, 256);
    k_warp<<<(20 * 64 * 128 + 255) / 256, 256, 0, stream>>>(feat1, pm, warp1, (_Float16*)0, 128, 64, 128);
    k_warp<<<(20 * 32 * 64 + 255) / 256, 256, 0, stream>>>(feat2, pm, warp2, (_Float16*)0, 256, 32, 64);

    int nblk = IMGS * HW0 / 64;   // 8192
    k_conv1_wmma<<<nblk, 128, 0, stream>>>(w0h, wp1, y1);
    k_instnorm<<<IMGS * OC1, 256, 0, stream>>>(y1, y1h, HW0, 1);
    k_conv2_wmma<<<nblk, 128, 0, stream>>>(y1h, wp2, y2);
    k_instnorm<<<IMGS * OC2, 256, 0, stream>>>(y2, (_Float16*)0, HW0, 1);
    k_conv3<<<(IMGS * HW0 + 255) / 256, 256, 0, stream>>>(y2, w3, y3);
    k_instnorm<<<IMGS * 8, 256, 0, stream>>>(y3, (_Float16*)0, HW0, 1);
    k_conv4<<<(IMGS * HW0 + 255) / 256, 256, 0, stream>>>(y3, w4, b4, g4);

    float* out0 = out;
    float* out1 = out0 + (size_t)B_ * 64 * 128 * 256;
    float* out2 = out1 + (size_t)B_ * 128 * 64 * 128;
    k_fuse<<<(B_ * 128 * 256 + 255) / 256, 256, 0, stream>>>(warp0, g4, out0, 64, 128, 256, 1);
    k_fuse<<<(B_ * 64 * 128 + 255) / 256, 256, 0, stream>>>(warp1, g4, out1, 128, 64, 128, 2);
    k_fuse<<<(B_ * 32 * 64 + 255) / 256, 256, 0, stream>>>(warp2, g4, out2, 256, 32, 64, 4);
}
